// GINEncoder_15049565405786
// MI455X (gfx1250) — matrix operand order repaired and checked by reading, not compiled, not run on hardware
//
#include <hip/hip_runtime.h>
#include <hip/hip_fp16.h>

typedef __attribute__((ext_vector_type(16))) _Float16 v16h;
typedef __attribute__((ext_vector_type(8)))  float    v8f;

#define N_ATOMS   50000
#define ATOM_FDIM 133
#define HIDDEN    128
#define DEPTH     3
#define N_EDGES   800000
#define N_MOLS    1024

__device__ __forceinline__ void atomAdd(float* p, float v) {
  // native GLOBAL_ATOMIC_ADD_F32 path (no CAS loop)
  unsafeAtomicAdd(p, v);
}

// ---------------------------------------------------------------------------
// Convert W1/W2 (fp32, [d][k][n]) to f16 stored K-contiguous: wt[(mat*3+d)][n][k]
// ---------------------------------------------------------------------------
__global__ void convw_kernel(const float* __restrict__ W1,
                             const float* __restrict__ W2,
                             _Float16* __restrict__ wt) {
  int idx = blockIdx.x * 256 + threadIdx.x;          // < 2*3*16384
  int k   = idx & 127;
  int n   = (idx >> 7) & 127;
  int d   = (idx >> 14) % 3;
  int mat = idx / (3 * 16384);
  const float* W = mat ? W2 : W1;
  wt[idx] = (_Float16)W[d * 16384 + k * 128 + n];
}

// ---------------------------------------------------------------------------
// x = relu(f_atoms @ W_in + b_in);  agg = (1+eps[0]) * x
// (K=133 -> scalar kernel; 16 atoms/block; W_in stays hot in L2)
// ---------------------------------------------------------------------------
__global__ __launch_bounds__(128) void input_proj_kernel(
    const float* __restrict__ f_atoms, const float* __restrict__ W_in,
    const float* __restrict__ b_in, const float* __restrict__ eps,
    float* __restrict__ x, float* __restrict__ agg) {
  __shared__ float F[16 * ATOM_FDIM];
  const int tid  = threadIdx.x;
  const int tile = blockIdx.x;
  const int base = tile * 16 * ATOM_FDIM;
  for (int idx = tid; idx < 16 * ATOM_FDIM; idx += 128) F[idx] = f_atoms[base + idx];
  __syncthreads();
  float acc[16];
#pragma unroll
  for (int a = 0; a < 16; ++a) acc[a] = 0.f;
  for (int k = 0; k < ATOM_FDIM; ++k) {
    float w = W_in[k * 128 + tid];
#pragma unroll
    for (int a = 0; a < 16; ++a) acc[a] = fmaf(F[a * ATOM_FDIM + k], w, acc[a]);
  }
  const float bias = b_in[tid];
  const float sc   = 1.0f + eps[0];
#pragma unroll
  for (int a = 0; a < 16; ++a) {
    float v = fmaxf(acc[a] + bias, 0.f);
    int   o = (tile * 16 + a) * 128 + tid;
    x[o]   = v;
    agg[o] = sc * v;
  }
}

// ---------------------------------------------------------------------------
// agg[tgt[e]] += x[src[e]]      (one wave32 per edge, float4 per lane)
// ---------------------------------------------------------------------------
__global__ __launch_bounds__(256) void edge_scatter_kernel(
    const float* __restrict__ x, const int* __restrict__ src,
    const int* __restrict__ tgt, float* __restrict__ agg) {
  int gw   = (blockIdx.x * 256 + threadIdx.x) >> 5;  // edge id
  int lane = threadIdx.x & 31;
  if (gw >= N_EDGES) return;
  int s = src[gw], t = tgt[gw];
  const float4 v = ((const float4*)x)[s * 32 + lane];
  float* dst = agg + t * 128 + lane * 4;
  atomAdd(dst + 0, v.x);
  atomAdd(dst + 1, v.y);
  atomAdd(dst + 2, v.z);
  atomAdd(dst + 3, v.w);
}

__global__ void bn_zero_kernel(float* bn) { bn[threadIdx.x] = 0.f; }  // 256 floats

// ---------------------------------------------------------------------------
// Fused: h1 = relu(agg@W1 + b1);  h2 = h1@W2 + b2;  bn_sum/bn_sq column stats.
// One block = 16 rows, 8 waves, each wave owns one 16-wide N-tile.
// WMMA f32_16x16x32_f16, K loop of 4.
// ---------------------------------------------------------------------------
__global__ __launch_bounds__(256) void fused_gemm_kernel(
    const float* __restrict__ agg,
    const _Float16* __restrict__ wt1,   // [n][k] f16 (K-contiguous)
    const _Float16* __restrict__ wt2,
    const float* __restrict__ b1, const float* __restrict__ b2,
    float* __restrict__ h2, float* __restrict__ bnsum, float* __restrict__ bnsq) {
  constexpr int AS = 136;                       // padded LDS row stride (halfs), 272 B
  __shared__ _Float16 Alds[16 * AS];
  __shared__ _Float16 Hlds[16 * AS];

  const int tid  = threadIdx.x;
  const int lane = tid & 31;
  const int wave = tid >> 5;                    // n-tile 0..7
  const int m    = lane & 15;
  const int hi   = lane >> 4;
  const int rowBase = blockIdx.x * 16;

  // Preload B operands (32x16 f16 chunks): lanes 0-15 K=0..15, lanes 16-31 K=16..31
  v16h B1[4], B2[4];
  {
    const int n  = wave * 16 + m;
    const int kb = hi * 16;
#pragma unroll
    for (int kk = 0; kk < 4; ++kk) {
      B1[kk] = *(const v16h*)(wt1 + n * 128 + kk * 32 + kb);
      B2[kk] = *(const v16h*)(wt2 + n * 128 + kk * 32 + kb);
    }
  }

  // Stage A tile: 16x128 fp32 -> f16 LDS, float4 loads + packed b64 LDS stores
#pragma unroll
  for (int i = 0; i < 2; ++i) {
    int idx = tid + i * 256;                    // float4 index, 0..511 (512 = 16*32)
    int r   = idx >> 5;                         // row (32 float4 per row)
    int c4  = idx & 31;
    float4 v = ((const float4*)agg)[(rowBase + r) * 32 + c4];
    union { _Float16 h[4]; unsigned long long u; } pk;
    pk.h[0] = (_Float16)v.x; pk.h[1] = (_Float16)v.y;
    pk.h[2] = (_Float16)v.z; pk.h[3] = (_Float16)v.w;
    *(unsigned long long*)(Alds + r * AS + c4 * 4) = pk.u;
  }
  __syncthreads();

  union AV { v16h h; unsigned int u[8]; };

  // GEMM1
  v8f acc = {};
#pragma unroll
  for (int kk = 0; kk < 4; ++kk) {
    AV a;
    const unsigned int* p0 = (const unsigned int*)(Alds + m * AS + kk * 32 + hi * 16);
    const unsigned int* p1 = (const unsigned int*)(Alds + m * AS + kk * 32 + hi * 16 + 8);
#pragma unroll
    for (int j = 0; j < 4; ++j) { a.u[j] = p0[j]; a.u[4 + j] = p1[j]; }
    acc = __builtin_amdgcn_wmma_f32_16x16x32_f16(false, a.h, false, B1[kk],
                                                 (short)0, acc, false, false);
  }
  {
    const float bias = b1[wave * 16 + m];       // column bias (n = col for this lane)
#pragma unroll
    for (int j = 0; j < 8; ++j) {
      float v = fmaxf(acc[j] + bias, 0.0f);     // relu(h1)
      Hlds[(hi * 8 + j) * AS + wave * 16 + m] = (_Float16)v;
    }
  }
  __syncthreads();

  // GEMM2
  v8f acc2 = {};
#pragma unroll
  for (int kk = 0; kk < 4; ++kk) {
    AV a;
    const unsigned int* p0 = (const unsigned int*)(Hlds + m * AS + kk * 32 + hi * 16);
    const unsigned int* p1 = (const unsigned int*)(Hlds + m * AS + kk * 32 + hi * 16 + 8);
#pragma unroll
    for (int j = 0; j < 4; ++j) { a.u[j] = p0[j]; a.u[4 + j] = p1[j]; }
    acc2 = __builtin_amdgcn_wmma_f32_16x16x32_f16(false, a.h, false, B2[kk],
                                                  (short)0, acc2, false, false);
  }

  const int col = wave * 16 + m;
  const float bias2 = b2[col];
  float s = 0.f, q = 0.f;
#pragma unroll
  for (int j = 0; j < 8; ++j) {
    float v = acc2[j] + bias2;
    h2[(rowBase + hi * 8 + j) * 128 + col] = v;
    s += v;  q += v * v;
  }
  // combine the two row-halves (same column) then one atomic per column per block
  s += __shfl_xor(s, 16, 32);
  q += __shfl_xor(q, 16, 32);
  if (hi == 0) {
    atomAdd(&bnsum[col], s);
    atomAdd(&bnsq[col], q);
  }
}

// ---------------------------------------------------------------------------
// x = relu(gamma*(h2-mean)*rsqrt(var+1e-5)+beta);
// optionally agg = (1+eps[d+1]) * x   (pre-init for next depth's scatter)
// ---------------------------------------------------------------------------
__global__ void bn_apply_kernel(const float* __restrict__ h2,
                                const float* __restrict__ bnsum,
                                const float* __restrict__ bnsq,
                                const float* __restrict__ gamma,
                                const float* __restrict__ beta,
                                const float* __restrict__ eps, int d,
                                int write_agg, float* __restrict__ x,
                                float* __restrict__ agg) {
  int i   = blockIdx.x * 256 + threadIdx.x;     // < N_ATOMS*128
  int col = i & 127;
  const float invN = 1.0f / (float)N_ATOMS;
  float mean = bnsum[col] * invN;
  float var  = bnsq[col] * invN - mean * mean;
  float inv  = rsqrtf(var + 1e-5f);
  float v = gamma[d * 128 + col] * (h2[i] - mean) * inv + beta[d * 128 + col];
  v = fmaxf(v, 0.f);
  x[i] = v;
  if (write_agg) {
    float sc = 1.0f + eps[d + 1];
    agg[i] = sc * v;
  }
}

// ---------------------------------------------------------------------------
// Pooling
// ---------------------------------------------------------------------------
__global__ void pool_zero_kernel(float* p) {    // molsum[1024*128] + molcnt[1024]
  p[blockIdx.x * 256 + threadIdx.x] = 0.f;
}

__global__ __launch_bounds__(256) void pool_scatter_kernel(
    const float* __restrict__ x, const int* __restrict__ seg,
    float* __restrict__ molsum, float* __restrict__ molcnt) {
  int gw   = (blockIdx.x * 256 + threadIdx.x) >> 5;   // atom id
  int lane = threadIdx.x & 31;
  if (gw >= N_ATOMS) return;
  int ms = seg[gw];
  const float4 v = ((const float4*)x)[gw * 32 + lane];
  float* dst = molsum + ms * 128 + lane * 4;
  atomAdd(dst + 0, v.x);
  atomAdd(dst + 1, v.y);
  atomAdd(dst + 2, v.z);
  atomAdd(dst + 3, v.w);
  if (lane == 0) atomAdd(&molcnt[ms], 1.0f);
}

__global__ void pool_final_kernel(const float* __restrict__ molsum,
                                  const float* __restrict__ molcnt,
                                  float* __restrict__ out) {
  int i  = blockIdx.x * 256 + threadIdx.x;      // < N_MOLS*128
  int ms = i >> 7;
  float c = molcnt[ms];
  out[i] = (c > 0.f) ? molsum[i] / c : 0.f;
}

// ---------------------------------------------------------------------------
extern "C" void kernel_launch(void* const* d_in, const int* in_sizes, int n_in,
                              void* d_out, int out_size, void* d_ws, size_t ws_size,
                              hipStream_t stream) {
  const float* f_atoms = (const float*)d_in[0];
  const float* W_in    = (const float*)d_in[1];
  const float* b_in    = (const float*)d_in[2];
  const float* W1      = (const float*)d_in[3];
  const float* b1      = (const float*)d_in[4];
  const float* W2      = (const float*)d_in[5];
  const float* b2      = (const float*)d_in[6];
  const float* gamma   = (const float*)d_in[7];
  const float* beta    = (const float*)d_in[8];
  const float* epsp    = (const float*)d_in[9];
  const int*   edge    = (const int*)d_in[10];
  const int*   seg     = (const int*)d_in[11];
  const int* src = edge;
  const int* tgt = edge + N_EDGES;

  char* ws = (char*)d_ws;
  const size_t XB = (size_t)N_ATOMS * 128 * 4;        // 25,600,000 B
  float*    x    = (float*)ws;
  float*    agg  = (float*)(ws + XB);
  float*    h2   = (float*)(ws + 2 * XB);
  _Float16* wt   = (_Float16*)(ws + 3 * XB);          // 2*3*16384 halfs = 196,608 B
  float*    bns  = (float*)(ws + 3 * XB + 196608);    // bnsum[128] + bnsq[128]
  float*    mols = (float*)(ws + 3 * XB + 196608 + 1024);   // molsum 524,288 B
  float*    molc = mols + N_MOLS * 128;               // molcnt 4,096 B
  float*    bnsum = bns;
  float*    bnsq  = bns + 128;

  // one-time per call: f16 transposed weights
  convw_kernel<<<(2 * 3 * 16384) / 256, 256, 0, stream>>>(W1, W2, wt);

  // input projection (also initializes agg for depth 0)
  input_proj_kernel<<<N_ATOMS / 16, 128, 0, stream>>>(f_atoms, W_in, b_in, epsp, x, agg);

  for (int d = 0; d < DEPTH; ++d) {
    bn_zero_kernel<<<1, 256, 0, stream>>>(bns);
    edge_scatter_kernel<<<(N_EDGES * 32) / 256, 256, 0, stream>>>(x, src, tgt, agg);
    fused_gemm_kernel<<<N_ATOMS / 16, 256, 0, stream>>>(
        agg, wt + d * 16384, wt + (3 + d) * 16384, b1 + d * 128, b2 + d * 128,
        h2, bnsum, bnsq);
    bn_apply_kernel<<<(N_ATOMS * 128) / 256, 256, 0, stream>>>(
        h2, bnsum, bnsq, gamma, beta, epsp, d, (d + 1 < DEPTH) ? 1 : 0, x, agg);
  }

  pool_zero_kernel<<<(N_MOLS * 128 + N_MOLS) / 256, 256, 0, stream>>>(mols);
  pool_scatter_kernel<<<(N_ATOMS * 32) / 256, 256, 0, stream>>>(x, seg, mols, molc);
  pool_final_kernel<<<(N_MOLS * 128) / 256, 256, 0, stream>>>(mols, molc, (float*)d_out);
}
